// RNNLayer_custom_11768210391742
// MI455X (gfx1250) — compile-verified, matrix operand order checked by reading, not run on hardware
//
#include <hip/hip_runtime.h>

typedef __attribute__((ext_vector_type(16))) _Float16 v16h;
typedef __attribute__((ext_vector_type(8)))  _Float16 v8h;
typedef __attribute__((ext_vector_type(4)))  _Float16 v4h;
typedef __attribute__((ext_vector_type(8)))  float    v8f;
typedef __attribute__((ext_vector_type(4)))  float    v4f;

#define SEQ    512
#define BATCH  256
#define IN_DIM 1024
#define HID    1024
#define GATES  (3 * HID)   // 3072

// ---------------------------------------------------------------------------
// f32 -> f16 elementwise conversion (vectorized x4, grid-stride) — weights only
// ---------------------------------------------------------------------------
__global__ void cvt_f32_to_f16(const float* __restrict__ src,
                               _Float16* __restrict__ dst,
                               long long n) {
    long long i      = (long long)blockIdx.x * blockDim.x + threadIdx.x;
    long long stride = (long long)gridDim.x * blockDim.x;
    for (long long j = i * 4; j < n; j += stride * 4) {
        v4f x = *(const v4f*)(src + j);
        v4h y;
        y.x = (_Float16)x.x; y.y = (_Float16)x.y;
        y.z = (_Float16)x.z; y.w = (_Float16)x.w;
        *(v4h*)(dst + j) = y;
    }
}

// ---------------------------------------------------------------------------
// Load one 16x32 f16 A-fragment from an f32 row pointer (row*K + kk + ksub),
// converting in-register. Halves 0..7 = K+0..7, halves 8..15 = K+16..23
// (lane<16) / K+8..15 and K+24..31 (lane>=16) — ksub already applied.
// ---------------------------------------------------------------------------
__device__ __forceinline__ v16h load_frag_f32(const float* __restrict__ p) {
    v4f x0 = *(const v4f*)(p);
    v4f x1 = *(const v4f*)(p + 4);
    v4f x2 = *(const v4f*)(p + 16);
    v4f x3 = *(const v4f*)(p + 20);
    v16h r;
    r[0]  = (_Float16)x0.x; r[1]  = (_Float16)x0.y;
    r[2]  = (_Float16)x0.z; r[3]  = (_Float16)x0.w;
    r[4]  = (_Float16)x1.x; r[5]  = (_Float16)x1.y;
    r[6]  = (_Float16)x1.z; r[7]  = (_Float16)x1.w;
    r[8]  = (_Float16)x2.x; r[9]  = (_Float16)x2.y;
    r[10] = (_Float16)x2.z; r[11] = (_Float16)x2.w;
    r[12] = (_Float16)x3.x; r[13] = (_Float16)x3.y;
    r[14] = (_Float16)x3.z; r[15] = (_Float16)x3.w;
    return r;
}

__device__ __forceinline__ v16h load_frag_f16(const _Float16* __restrict__ p) {
    v8h lo = *(const v8h*)(p);
    v8h hi = *(const v8h*)(p + 16);
    return __builtin_shufflevector(lo, hi, 0,1,2,3,4,5,6,7,8,9,10,11,12,13,14,15);
}

// ---------------------------------------------------------------------------
// Fused dual GEMM for one GRU timestep (one launch, blockIdx.z selects):
//   z=0: G1[256,3072] = x_t[256,1024](f32) @ Wih16[3072,1024]^T + bih
//   z=1: G2[256,3072] = h_{t-1}[256,1024](f32) @ Whh16[3072,1024]^T + bhh
// A is f32, converted to f16 in-register (cvt co-executes with XDL WMMA).
// Block = 256 threads = 8 waves (2 in M x 4 in N); wave tile = 32x64
// (2x4 of 16x16 WMMA tiles, 8 WMMAs per 6 fragment loads per K-step).
// Grid = (3072/256, 256/64, 2) = (12, 4, 2).
// ---------------------------------------------------------------------------
__global__ __launch_bounds__(256) void gru_dual_gemm_wmma(
    const float*    __restrict__ xA,     // [256 x 1024] f32 (x_t)
    const float*    __restrict__ hA,     // [256 x 1024] f32 (h_{t-1})
    const _Float16* __restrict__ wih16,  // [3072 x 1024] f16
    const _Float16* __restrict__ whh16,  // [3072 x 1024] f16
    const float*    __restrict__ bih,    // [3072]
    const float*    __restrict__ bhh,    // [3072]
    float*          __restrict__ G1,     // [256 x 3072]
    float*          __restrict__ G2)     // [256 x 3072]
{
    const int K = 1024;
    const bool hz = (blockIdx.z != 0);                 // uniform select
    const float*    A    = hz ? hA    : xA;
    const _Float16* W    = hz ? whh16 : wih16;
    const float*    bias = hz ? bhh   : bih;
    float*          OUT  = hz ? G2    : G1;

    const int lane = threadIdx.x & 31;
    const int wave = threadIdx.x >> 5;   // 0..7
    const int wm   = wave >> 2;          // 0..1 (M)
    const int wn   = wave & 3;           // 0..3 (N)
    const int m0   = blockIdx.y * 64  + wm * 32;
    const int n0   = blockIdx.x * 256 + wn * 64;

    const int lrow = lane & 15;
    const int ksub = (lane < 16) ? 0 : 8;

    // A rows (f32), two fragments (M and M+16)
    const float* pa0 = A + (size_t)(m0 + lrow)      * K + ksub;
    const float* pa1 = A + (size_t)(m0 + 16 + lrow) * K + ksub;
    // B columns from row-major [N,K] weights (f16), four fragments
    const _Float16* pb0 = W + (size_t)(n0 +  0 + lrow) * K + ksub;
    const _Float16* pb1 = W + (size_t)(n0 + 16 + lrow) * K + ksub;
    const _Float16* pb2 = W + (size_t)(n0 + 32 + lrow) * K + ksub;
    const _Float16* pb3 = W + (size_t)(n0 + 48 + lrow) * K + ksub;

    v8f acc00 = {}; v8f acc01 = {}; v8f acc02 = {}; v8f acc03 = {};
    v8f acc10 = {}; v8f acc11 = {}; v8f acc12 = {}; v8f acc13 = {};

#pragma unroll 1
    for (int kk = 0; kk < K; kk += 32) {
        // stream next weight block through cache (global_prefetch_b8)
        __builtin_prefetch(pb0 + kk + 128, 0, 1);
        __builtin_prefetch(pb2 + kk + 128, 0, 1);

        v16h a0 = load_frag_f32(pa0 + kk);
        v16h a1 = load_frag_f32(pa1 + kk);
        v16h b0 = load_frag_f16(pb0 + kk);
        v16h b1 = load_frag_f16(pb1 + kk);
        v16h b2 = load_frag_f16(pb2 + kk);
        v16h b3 = load_frag_f16(pb3 + kk);

        acc00 = __builtin_amdgcn_wmma_f32_16x16x32_f16(false, a0, false, b0, (short)0, acc00, false, false);
        acc01 = __builtin_amdgcn_wmma_f32_16x16x32_f16(false, a0, false, b1, (short)0, acc01, false, false);
        acc02 = __builtin_amdgcn_wmma_f32_16x16x32_f16(false, a0, false, b2, (short)0, acc02, false, false);
        acc03 = __builtin_amdgcn_wmma_f32_16x16x32_f16(false, a0, false, b3, (short)0, acc03, false, false);
        acc10 = __builtin_amdgcn_wmma_f32_16x16x32_f16(false, a1, false, b0, (short)0, acc10, false, false);
        acc11 = __builtin_amdgcn_wmma_f32_16x16x32_f16(false, a1, false, b1, (short)0, acc11, false, false);
        acc12 = __builtin_amdgcn_wmma_f32_16x16x32_f16(false, a1, false, b2, (short)0, acc12, false, false);
        acc13 = __builtin_amdgcn_wmma_f32_16x16x32_f16(false, a1, false, b3, (short)0, acc13, false, false);
    }

    // Epilogue. C/D 16x16 f32 layout: lane(0-15)=col, VGPR v = row m0+v
    // (lanes 0-15) or m0+8+v (lanes 16-31).
    const int c0 = n0 + lrow;
    const float bz0 = bias[c0];
    const float bz1 = bias[c0 + 16];
    const float bz2 = bias[c0 + 32];
    const float bz3 = bias[c0 + 48];
    const int r0 = m0 + ((lane < 16) ? 0 : 8);
    const int r1 = r0 + 16;

#pragma unroll
    for (int v = 0; v < 8; ++v) {
        float* o0 = OUT + (size_t)(r0 + v) * GATES + c0;
        float* o1 = OUT + (size_t)(r1 + v) * GATES + c0;
        o0[0]  = acc00[v] + bz0;
        o0[16] = acc01[v] + bz1;
        o0[32] = acc02[v] + bz2;
        o0[48] = acc03[v] + bz3;
        o1[0]  = acc10[v] + bz0;
        o1[16] = acc11[v] + bz1;
        o1[32] = acc12[v] + bz2;
        o1[48] = acc13[v] + bz3;
    }
}

// ---------------------------------------------------------------------------
// GRU elementwise update for one timestep (vectorized x4).
// n-gate uses separate input/hidden pre-activations: tanh(ni + r * nh).
// ---------------------------------------------------------------------------
__global__ void gru_update(const float* __restrict__ G1,
                           const float* __restrict__ G2,
                           const float* __restrict__ hprev,
                           float*       __restrict__ out_t)
{
    const int i4 = (blockIdx.x * blockDim.x + threadIdx.x) * 4;  // 0..B*H-1
    const int b  = i4 >> 10;        // / HID
    const int j  = i4 & (HID - 1);

    const float* g1 = G1 + (size_t)b * GATES;
    const float* g2 = G2 + (size_t)b * GATES;

    v4f ri = *(const v4f*)(g1 + j);
    v4f ui = *(const v4f*)(g1 + HID + j);
    v4f ni = *(const v4f*)(g1 + 2 * HID + j);
    v4f rh = *(const v4f*)(g2 + j);
    v4f uh = *(const v4f*)(g2 + HID + j);
    v4f nh = *(const v4f*)(g2 + 2 * HID + j);
    v4f h  = *(const v4f*)(hprev + i4);

    v4f hy;
#pragma unroll
    for (int c = 0; c < 4; ++c) {
        float r  = 1.0f / (1.0f + __expf(-(ri[c] + rh[c])));
        float u  = 1.0f / (1.0f + __expf(-(ui[c] + uh[c])));
        float nn = tanhf(ni[c] + r * nh[c]);
        hy[c]    = u * h[c] + (1.0f - u) * nn;
    }
    *(v4f*)(out_t + i4) = hy;
}

// ---------------------------------------------------------------------------
// Launch
// inputs: [0]=input [512,256,1024] f32, [1]=state [1,256,1024] f32,
//         [2]=weight_ih [3072,1024] f32, [3]=weight_hh [3072,1024] f32,
//         [4]=bias_ih [3072] f32, [5]=bias_hh [3072] f32
// output: [512,256,1024] f32
// ---------------------------------------------------------------------------
extern "C" void kernel_launch(void* const* d_in, const int* in_sizes, int n_in,
                              void* d_out, int out_size, void* d_ws, size_t ws_size,
                              hipStream_t stream) {
    const float* input = (const float*)d_in[0];
    const float* state = (const float*)d_in[1];
    const float* wih   = (const float*)d_in[2];
    const float* whh   = (const float*)d_in[3];
    const float* bih   = (const float*)d_in[4];
    const float* bhh   = (const float*)d_in[5];
    float* out = (float*)d_out;

    char* ws = (char*)d_ws;
    _Float16* wih16 = (_Float16*)ws;  ws += (size_t)GATES * IN_DIM * 2;
    _Float16* whh16 = (_Float16*)ws;  ws += (size_t)GATES * HID * 2;
    float*    G1b   = (float*)ws;     ws += (size_t)BATCH * GATES * 4;
    float*    G2b   = (float*)ws;

    // One-time weight conversion; f16 weights (12 MB) stay L2-resident.
    cvt_f32_to_f16<<<512, 256, 0, stream>>>(wih, wih16, (long long)GATES * IN_DIM);
    cvt_f32_to_f16<<<512, 256, 0, stream>>>(whh, whh16, (long long)GATES * HID);

    const dim3 ggrid(GATES / 256, BATCH / 64, 2);  // (12, 4, 2)
    for (int t = 0; t < SEQ; ++t) {
        const float* xt    = input + (size_t)t * BATCH * IN_DIM;
        const float* hprev = (t == 0) ? state : (out + (size_t)(t - 1) * BATCH * HID);
        gru_dual_gemm_wmma<<<ggrid, 256, 0, stream>>>(
            xt, hprev, wih16, whh16, bih, bhh, G1b, G2b);
        gru_update<<<(BATCH * HID) / 1024, 256, 0, stream>>>(
            G1b, G2b, hprev, out + (size_t)t * BATCH * HID);
    }
}
